// GCN_2LSTM_16166256902761
// MI455X (gfx1250) — matrix operand order repaired and checked by reading, not compiled, not run on hardware
//
#include <hip/hip_runtime.h>
#include <cstdint>
#include <cstddef>

// ---------------------------------------------------------------------------
// MI455X (gfx1250) implementation of the 2-level GCN-LSTM reference.
//
// Design (reasoned from MI455X specs, compile-only):
//  * Dense h@W GEMMs (~100 GFLOP) -> v_wmma_f32_16x16x32_f16, A/B in f16
//    (activations are tanh-bounded), f32 accumulate. One 16x16 D-tile/wave.
//  * Per-layer weights (<=112KB f16) pre-swizzled into WMMA B-fragment order,
//    staged whole into LDS (320KB/WGP) by the Tensor Data Mover as a linear
//    1D tile (tensor_load_to_lds + s_wait_tensorcnt). B-fragment reads are
//    then 2x ds_load_b128 per lane; A-fragment reads are 2x global_load_b128.
//  * Edge aggregation (~27GB gather/scatter) is the true bottleneck; hw
//    matrices (<=51MB) stay L2-resident (192MB L2), scatter uses f32 atomics.
//  * segment_max(batch) then max over graphs == global column max -> one
//    column-reduction kernel; `batch` input is mathematically irrelevant.
// ---------------------------------------------------------------------------

typedef __attribute__((ext_vector_type(16))) _Float16 v16h;
typedef __attribute__((ext_vector_type(8)))  _Float16 v8h;
typedef __attribute__((ext_vector_type(8)))  float    v8f;
typedef __attribute__((ext_vector_type(4)))  unsigned u32x4;
typedef __attribute__((ext_vector_type(8)))  int      i32x8;
typedef __attribute__((ext_vector_type(4)))  int      i32x4;

#ifndef __has_builtin
#define __has_builtin(x) 0
#endif
#if defined(__HIP_DEVICE_COMPILE__) && __has_builtin(__builtin_amdgcn_tensor_load_to_lds)
#define ATHENA_TDM 1
#endif

// Activation codes for the fused GCN finalize.
#define ACT_NONE      0
#define ACT_TANH      1
#define ACT_SIGMOID   2
#define ACT_SIG_TANH  3   // sigmoid(tanh(x)) : sigmoid applied on inception output
#define ACT_TANH_TANH 4   // tanh(tanh(x))    : tanh applied on inception output

__device__ __forceinline__ float actf(float x, int a) {
  switch (a) {
    case ACT_TANH:      return tanhf(x);
    case ACT_SIGMOID:   return 1.0f / (1.0f + expf(-x));
    case ACT_SIG_TANH:  { float t = tanhf(x); return 1.0f / (1.0f + expf(-t)); }
    case ACT_TANH_TANH: { float t = tanhf(x); return tanhf(t); }
    default:            return x;
  }
}

// ---------------------------------------------------------------------------
// Degree + symmetric-norm vectors
// ---------------------------------------------------------------------------
__global__ void k_deg(const int* __restrict__ dst, float* __restrict__ deg, int nE) {
  int e = blockIdx.x * 256 + threadIdx.x;
  if (e < nE) atomicAdd(&deg[dst[e]], 1.0f);
}

__global__ void k_dn(const float* __restrict__ deg, float* __restrict__ dn1,
                     float* __restrict__ dn2, int n, int nPad) {
  int i = blockIdx.x * 256 + threadIdx.x;
  if (i >= nPad) return;
  if (i < n) {
    float d = deg[i];
    dn1[i] = rsqrtf(d + 1.0f);
    dn2[i] = rsqrtf(d + 2.0f);
  } else {
    dn1[i] = 0.0f; dn2[i] = 0.0f;
  }
}

// ---------------------------------------------------------------------------
// f32 -> f16 activation stager: 8 halves (one b128 store) per thread,
// zero-padded to [nPad x Kpad].
// ---------------------------------------------------------------------------
__global__ void k_tohalf(const float* __restrict__ X, int xstride, int K,
                         _Float16* __restrict__ A16, int Kpad, int n, int nPad) {
  int idx = blockIdx.x * 256 + threadIdx.x;
  int kchunks = Kpad >> 3;
  int total = nPad * kchunks;
  if (idx >= total) return;
  int i  = idx / kchunks;
  int k0 = (idx - i * kchunks) * 8;
  v8h o;
#pragma unroll
  for (int j = 0; j < 8; ++j) {
    int k = k0 + j;
    float v = (i < n && k < K) ? X[(size_t)i * xstride + k] : 0.0f;
    o[j] = (_Float16)v;
  }
  *(v8h*)(A16 + (size_t)i * Kpad + k0) = o;
}

// ---------------------------------------------------------------------------
// Weight converter: f32 [K x M] -> f16 blob in WMMA *B-fragment order*:
//   blob[(((ct*nKT + ktile)*32 + lane)*16) + j]  holds  W[k, m]
//   with kin = 8*(lane>>4) + (j&7) + 16*(j>>3),  k = ktile*32 + kin,  m = ct*16 + (lane&15).
// Each lane's 16 B-operand halves are then contiguous (2x ds_load_b128).
// ---------------------------------------------------------------------------
__global__ void k_wcvt(const float* __restrict__ W, _Float16* __restrict__ W16,
                       int K, int M, int Kpad, int Mpad) {
  int idx = blockIdx.x * 256 + threadIdx.x;
  int total = Kpad * Mpad;
  if (idx >= total) return;
  int nKT = Kpad >> 5;
  int j     = idx & 15;
  int lane  = (idx >> 4) & 31;
  int t     = idx >> 9;
  int ktile = t % nKT;
  int ct    = t / nKT;
  int k = ktile * 32 + 8 * (lane >> 4) + (j & 7) + 16 * (j >> 3);
  int m = ct * 16 + (lane & 15);
  float v = (k < K && m < M) ? W[(size_t)k * M + m] : 0.0f;
  W16[idx] = (_Float16)v;
}

// ---------------------------------------------------------------------------
// WMMA GEMM: HW[nPad x Mpad] = A16[nPad x Kpad] * W (pre-swizzled f16 blob)
// 8 waves/block, one 16x16 D-tile per wave per column-tile iteration.
// Whole swizzled W staged to LDS (max 112KB) via one 1D TDM descriptor.
// ---------------------------------------------------------------------------
#define GEMM_WAVES 8

__global__ void __launch_bounds__(256) k_gemm(const _Float16* __restrict__ A,
                                              const _Float16* __restrict__ Wg,
                                              float* __restrict__ C,
                                              int nPad, int Kpad, int Mpad) {
  __shared__ _Float16 sW[57344];  // 112 KB: fits 224x256 (largest layer)
  const int total = Kpad * Mpad;

#if defined(ATHENA_TDM)
  if (threadIdx.x < 32) {  // wave 0 issues one 1D TDM descriptor for all of W
    unsigned lds = (unsigned)(uintptr_t)(void*)&sW[0];      // low 32b = LDS byte offset
    uint64_t ga  = (uint64_t)(uintptr_t)Wg;
    u32x4 g0;
    g0[0] = 1u;                                             // count=1, user descriptor
    g0[1] = lds;                                            // lds_addr
    g0[2] = (unsigned)(ga & 0xFFFFFFFFu);                   // global_addr[31:0]
    g0[3] = (unsigned)((ga >> 32) & 0x01FFFFFFu) | 0x80000000u; // addr[56:32] | type=2
    i32x8 g1;
    g1[0] = (int)(1u << 16);                                // data_size = 2 bytes
    g1[1] = (int)(((unsigned)total & 0xFFFFu) << 16);       // tensor_dim0 lo16
    g1[2] = (int)(((unsigned)total >> 16) | (1u << 16));    // dim0 hi16 | tensor_dim1 = 1
    g1[3] = (int)(((unsigned)total & 0xFFFFu) << 16);       // dim1 hi | tile_dim0 = total
    g1[4] = 1;                                              // tile_dim1 = 1 (tile_dim2 = 0)
    g1[5] = (int)(unsigned)total;                           // tensor_dim0_stride lo32
    g1[6] = 0;
    g1[7] = 0;
    i32x4 gz = {0, 0, 0, 0};                                // <=2D: groups 2/3 unused
#if defined(__clang_major__) && __clang_major__ >= 23
    i32x8 gz8 = {0, 0, 0, 0, 0, 0, 0, 0};
    __builtin_amdgcn_tensor_load_to_lds(g0, g1, gz, gz, gz8, 0);
#else
    __builtin_amdgcn_tensor_load_to_lds(g0, g1, gz, gz, 0);
#endif
#if __has_builtin(__builtin_amdgcn_s_wait_tensorcnt)
    __builtin_amdgcn_s_wait_tensorcnt(0);
#else
    asm volatile("s_wait_tensorcnt 0" ::: "memory");
#endif
  }
#else
  // Fallback: cooperative global->LDS copy (b128 chunks).
  for (int i = threadIdx.x; i < (total >> 3); i += 256)
    ((v8h*)sW)[i] = ((const v8h*)Wg)[i];
#endif
  __syncthreads();

  const int lane = threadIdx.x & 31;
  const int wave = threadIdx.x >> 5;
  const int rt   = blockIdx.x * GEMM_WAVES + wave;  // 16-row tile index
  if (rt * 16 >= nPad) return;

  const int rlo = lane & 15;
  const int kbA = (lane >> 4) * 8;      // A 16-bit layout: lanes16-31 start at K+8
  const int nKT = Kpad >> 5;
  const _Float16* __restrict__ arow = A + (size_t)(rt * 16 + rlo) * Kpad;

  const int nCT = Mpad >> 4;
  for (int ct = 0; ct < nCT; ++ct) {
    v8f acc = {};
    const _Float16* __restrict__ bbase = sW + ((size_t)(ct * nKT) * 32 + lane) * 16;
    for (int kt = 0; kt < nKT; ++kt) {
      __builtin_prefetch(arow + kt * 32 + 128, 0, 1);   // global_prefetch_b8
      // A fragment: two contiguous 16B runs per lane -> 2x global_load_b128
      v8h alo = *(const v8h*)(arow + kt * 32 + kbA);
      v8h ahi = *(const v8h*)(arow + kt * 32 + 16 + kbA);
      v16h a = __builtin_shufflevector(alo, ahi, 0, 1, 2, 3, 4, 5, 6, 7,
                                       8, 9, 10, 11, 12, 13, 14, 15);
      // B fragment: 32 contiguous bytes per lane -> 2x ds_load_b128
      const _Float16* bp = bbase + (size_t)kt * 512;
      v8h blo = *(const v8h*)(bp);
      v8h bhi = *(const v8h*)(bp + 8);
      v16h b = __builtin_shufflevector(blo, bhi, 0, 1, 2, 3, 4, 5, 6, 7,
                                       8, 9, 10, 11, 12, 13, 14, 15);
      acc = __builtin_amdgcn_wmma_f32_16x16x32_f16(
          /*neg_a=*/false, a, /*neg_b=*/false, b,
          /*c_mod=*/(short)0, acc, /*reuse_a=*/false, /*reuse_b=*/false);
    }
    const int m    = ct * 16 + rlo;
    const int row0 = rt * 16 + 8 * (lane >> 4);   // C/D: VGPR r -> M = r (+8 hi lanes)
#pragma unroll
    for (int r = 0; r < 8; ++r)
      C[(size_t)(row0 + r) * Mpad + m] = acc[r];
  }
}

// ---------------------------------------------------------------------------
// Edge scatter: agg[dst] += hw[src] * dn[src]*dn[dst]   (float4 per thread)
// ---------------------------------------------------------------------------
__global__ void k_scatter(const float* __restrict__ hw, const int* __restrict__ src,
                          const int* __restrict__ dst, const float* __restrict__ dn,
                          float* __restrict__ agg, int Mpad, int nE) {
  int idx = blockIdx.x * 256 + threadIdx.x;
  int chunks = Mpad >> 2;
  int e = idx / chunks;
  if (e >= nE) return;
  int c = (idx - e * chunks) * 4;
  int s = src[e], d = dst[e];
  float coef = dn[s] * dn[d];
  const float4 v = *(const float4*)(hw + (size_t)s * Mpad + c);
  float* o = agg + (size_t)d * Mpad + c;
  atomicAdd(o + 0, v.x * coef);
  atomicAdd(o + 1, v.y * coef);
  atomicAdd(o + 2, v.z * coef);
  atomicAdd(o + 3, v.w * coef);
}

// ---------------------------------------------------------------------------
// Finalize: out = act(agg + fill*dn^2*hw + b); padded columns forced to 0.
// ---------------------------------------------------------------------------
__global__ void k_finalize(const float* __restrict__ agg, const float* __restrict__ hw,
                           const float* __restrict__ dn, const float* __restrict__ bias,
                           float fill, int act, float* __restrict__ out,
                           int n, int M, int Mpad) {
  int idx = blockIdx.x * 256 + threadIdx.x;
  int i = idx / Mpad;
  if (i >= n) return;
  int f = idx - i * Mpad;
  float r = 0.0f;
  if (f < M) {
    float d = dn[i];
    float v = agg[(size_t)i * Mpad + f] + fill * d * d * hw[(size_t)i * Mpad + f] + bias[f];
    r = actf(v, act);
  }
  out[(size_t)i * Mpad + f] = r;
}

// ---------------------------------------------------------------------------
// Column max over all real nodes (== segment_max then max over graphs).
// One block per feature column.
// ---------------------------------------------------------------------------
__global__ void k_colmax(const float* __restrict__ X, int stride, int n,
                         float* __restrict__ gmax) {
  __shared__ float red[256];
  int c = blockIdx.x;
  float m = -3.402823e38f;
  for (int i = threadIdx.x; i < n; i += 256) m = fmaxf(m, X[(size_t)i * stride + c]);
  red[threadIdx.x] = m;
  __syncthreads();
  for (int s = 128; s > 0; s >>= 1) {
    if (threadIdx.x < s) red[threadIdx.x] = fmaxf(red[threadIdx.x], red[threadIdx.x + s]);
    __syncthreads();
  }
  if (threadIdx.x == 0) gmax[c] = red[0];
}

// ---------------------------------------------------------------------------
// Glue kernels: broadcast row, column-slab copy, elementwise mul / mad, strip.
// ---------------------------------------------------------------------------
__global__ void k_bcast(const float* __restrict__ row, float* __restrict__ Y,
                        int Ys, int off, int w, int n) {
  int idx = blockIdx.x * 256 + threadIdx.x;
  int i = idx / w;
  if (i >= n) return;
  int c = idx - i * w;
  Y[(size_t)i * Ys + off + c] = row[c];
}

__global__ void k_copycols(const float* __restrict__ X, int Xs, float* __restrict__ Y,
                           int Ys, int off, int w, int n) {
  int idx = blockIdx.x * 256 + threadIdx.x;
  int i = idx / w;
  if (i >= n) return;
  int c = idx - i * w;
  Y[(size_t)i * Ys + off + c] = X[(size_t)i * Xs + c];
}

__global__ void k_ewmul(const float* __restrict__ A, const float* __restrict__ B,
                        float* __restrict__ O, int w, int n) {
  int idx = blockIdx.x * 256 + threadIdx.x;
  if (idx >= n * w) return;
  O[idx] = A[idx] * B[idx];
}

__global__ void k_ewmad(const float* __restrict__ A, const float* __restrict__ B,
                        const float* __restrict__ Cc, float* __restrict__ O, int w, int n) {
  int idx = blockIdx.x * 256 + threadIdx.x;
  if (idx >= n * w) return;
  O[idx] = fmaf(A[idx], B[idx], Cc[idx]);
}

__global__ void k_strip(const float* __restrict__ X, int Xs, float* __restrict__ out,
                        int n, int M) {
  int idx = blockIdx.x * 256 + threadIdx.x;
  int i = idx / M;
  if (i >= n) return;
  int c = idx - i * M;
  out[idx] = X[(size_t)i * Xs + c];
}

// ---------------------------------------------------------------------------
// Host-side orchestration
// ---------------------------------------------------------------------------
namespace {

struct GcnP  { const float* W; const float* b; int K; int M; };
struct IncP  { GcnP c1, c2, c3, c4, c7; };
struct LstmP { GcnP conv1, conv2; IncP inc[6]; };

struct Env {
  hipStream_t st;
  int n, nPad, nE;
  const float* x;
  const int *srcI, *dstI;
  float *deg, *dn1, *dn2, *gmax;
  float *A, *B, *C, *D, *E;       // gcn_lstm block vars (width <= 64)
  float *P1, *P2;                  // wide temps / final-chain ping-pong (256)
  float *CAT;                      // concat buffer (576)
  float *HW, *AGG;                 // per-GCN transients (256)
  float *HOUT;                     // inter-block activations (192)
  _Float16 *A16, *W16;
};

inline int pad16(int x) { return (x + 15) & ~15; }
inline int pad32(int x) { return (x + 31) & ~31; }
inline int gcount(long total) { return (int)((total + 255) / 256); }

// One GCN layer: hw = h@W ; agg = scatter(hw) ; out = act(agg + fill*dn^2*hw + b)
void gcn_layer(Env& V, const float* hin, int hstride, int K, const GcnP& g,
               const float* dn, float fill, int act, float* out) {
  const int Kpad = pad32(K), Mpad = pad16(g.M);
  k_wcvt<<<gcount((long)Kpad * Mpad), 256, 0, V.st>>>(g.W, V.W16, K, g.M, Kpad, Mpad);
  k_tohalf<<<gcount((long)V.nPad * (Kpad >> 3)), 256, 0, V.st>>>(
      hin, hstride, K, V.A16, Kpad, V.n, V.nPad);
  k_gemm<<<(V.nPad / 16 + GEMM_WAVES - 1) / GEMM_WAVES, 256, 0, V.st>>>(
      V.A16, V.W16, V.HW, V.nPad, Kpad, Mpad);
  hipMemsetAsync(V.AGG, 0, (size_t)V.nPad * Mpad * sizeof(float), V.st);
  k_scatter<<<gcount((long)V.nE * (Mpad >> 2)), 256, 0, V.st>>>(
      V.HW, V.srcI, V.dstI, dn, V.AGG, Mpad, V.nE);
  k_finalize<<<gcount((long)V.n * Mpad), 256, 0, V.st>>>(
      V.AGG, V.HW, dn, g.b, fill, act, out, V.n, g.M, Mpad);
}

// Inception block (input width d, output width d); finalAct folds the caller's
// extra sigmoid/tanh into the c7 finalize.
void inception(Env& V, const float* h, int d, const IncP& p, int finalAct, float* out) {
  const int d2 = 2 * d, d4 = 4 * d, d9 = 9 * d;
  gcn_layer(V, h, d, d, p.c1, V.dn1, 1.0f, ACT_TANH, V.P1);           // [n,2d]
  gcn_layer(V, V.P1, d2, d2, p.c2, V.dn1, 1.0f, ACT_TANH, V.P2);      // [n,4d]
  k_colmax<<<d4, 256, 0, V.st>>>(V.P2, d4, V.n, V.gmax);              // pool
  gcn_layer(V, h, d, d, p.c3, V.dn1, 1.0f, ACT_TANH, V.P1);           // [n,2d]
  gcn_layer(V, V.P1, d2, d2, p.c4, V.dn1, 1.0f, ACT_TANH, V.P2);      // [n,4d]
  // CAT = [ broadcast(colmax)  |  i2  |  h ]   width 9d (288 or 576)
  k_bcast<<<gcount((long)V.n * d4), 256, 0, V.st>>>(V.gmax, V.CAT, d9, 0, d4, V.n);
  k_copycols<<<gcount((long)V.n * d4), 256, 0, V.st>>>(V.P2, d4, V.CAT, d9, d4, d4, V.n);
  k_copycols<<<gcount((long)V.n * d), 256, 0, V.st>>>(h, d, V.CAT, d9, 2 * d4, d, V.n);
  gcn_layer(V, V.CAT, d9, d9, p.c7, V.dn1, 1.0f, finalAct, out);      // [n,d]
}

void gcn_lstm(Env& V, const float* hin, int hstride, int din, const LstmP& p,
              int op, int dout, float* out) {
  gcn_layer(V, hin, hstride, din, p.conv1, V.dn1, 1.0f, ACT_NONE, V.A);  // h
  inception(V, V.A, op, p.inc[0], ACT_TANH, V.B);
  inception(V, V.B, op, p.inc[1], ACT_TANH, V.C);
  k_ewmul<<<gcount((long)V.n * op), 256, 0, V.st>>>(V.A, V.C, V.B, op, V.n);  // f = h*f
  inception(V, V.A, op, p.inc[2], ACT_TANH, V.C);
  inception(V, V.C, op, p.inc[3], ACT_SIG_TANH, V.D);                         // i1
  inception(V, V.A, op, p.inc[4], ACT_TANH, V.C);
  inception(V, V.C, op, p.inc[5], ACT_TANH_TANH, V.E);                        // i2
  k_ewmad<<<gcount((long)V.n * op), 256, 0, V.st>>>(V.D, V.E, V.B, V.C, op, V.n); // i
  gcn_layer(V, V.C, op, op, p.conv2, V.dn1, 1.0f, ACT_TANH, out);
}

}  // namespace

extern "C" void kernel_launch(void* const* d_in, const int* in_sizes, int n_in,
                              void* d_out, int out_size, void* d_ws, size_t ws_size,
                              hipStream_t stream) {
  Env V{};
  V.st   = stream;
  V.n    = in_sizes[0] / 3;                       // 50000
  V.nE   = in_sizes[1];                           // 800000
  V.nPad = ((V.n + 127) / 128) * 128;
  V.x    = (const float*)d_in[0];
  V.srcI = (const int*)d_in[1];
  V.dstI = (const int*)d_in[2];
  // d_in[3] = batch : segment_max then max over graphs == global max -> unused.

  // ---- walk the flattened params (python dict insertion order) ----
  int cur = 4;
  auto take = [&](int K, int M) -> GcnP {
    GcnP g; g.W = (const float*)d_in[cur++]; g.b = (const float*)d_in[cur++];
    g.K = K; g.M = M; return g;
  };
  auto takeInc = [&](int d) -> IncP {
    IncP p;
    p.c1 = take(d, 2 * d); p.c2 = take(2 * d, 4 * d);
    p.c3 = take(d, 2 * d); p.c4 = take(2 * d, 4 * d);
    p.c7 = take(9 * d, d);
    return p;
  };
  auto takeLstm = [&](int din, int dout, int op) -> LstmP {
    LstmP p;
    p.conv1 = take(din, op);
    p.conv2 = take(op, dout);
    for (int i = 0; i < 6; ++i) p.inc[i] = takeInc(op);
    return p;
  };
  LstmP L1 = takeLstm(3, 64, 32);
  LstmP L2 = takeLstm(64, 192, 64);
  GcnP pc1 = take(192 + 3, 256), pc2 = take(256, 128), pc3 = take(128, 64),
       pc4 = take(64, 32), pco = take(32, 3);

  // ---- carve workspace (bump allocator, 256B aligned) ----
  size_t off = 0;
  auto alloc = [&](size_t bytes) -> void* {
    void* p = (char*)d_ws + off;
    off = (off + bytes + 255) & ~(size_t)255;
    return p;
  };
  const size_t nP = (size_t)V.nPad;
  V.deg  = (float*)alloc(nP * 4);
  V.dn1  = (float*)alloc(nP * 4);
  V.dn2  = (float*)alloc(nP * 4);
  V.gmax = (float*)alloc(256 * 4);
  V.A    = (float*)alloc(nP * 64 * 4);
  V.B    = (float*)alloc(nP * 64 * 4);
  V.C    = (float*)alloc(nP * 64 * 4);
  V.D    = (float*)alloc(nP * 64 * 4);
  V.E    = (float*)alloc(nP * 64 * 4);
  V.P1   = (float*)alloc(nP * 256 * 4);
  V.P2   = (float*)alloc(nP * 256 * 4);
  V.CAT  = (float*)alloc(nP * 576 * 4);
  V.HW   = (float*)alloc(nP * 256 * 4);
  V.AGG  = (float*)alloc(nP * 256 * 4);
  V.HOUT = (float*)alloc(nP * 192 * 4);
  V.A16  = (_Float16*)alloc(nP * 576 * 2);
  V.W16  = (_Float16*)alloc(57600 * 2);
  (void)ws_size; (void)n_in; (void)out_size;

  // ---- degree + GCN norms ----
  hipMemsetAsync(V.deg, 0, nP * 4, stream);
  k_deg<<<gcount(V.nE), 256, 0, stream>>>(V.dstI, V.deg, V.nE);
  k_dn<<<gcount(V.nPad), 256, 0, stream>>>(V.deg, V.dn1, V.dn2, V.n, V.nPad);

  // ---- two GCN-LSTM blocks ----
  gcn_lstm(V, V.x, 3, 3, L1, 32, 64, V.HOUT);          // [n,64]  (stride 64)
  gcn_lstm(V, V.HOUT, 64, 64, L2, 64, 192, V.HOUT);    // [n,192] (input consumed first)

  // ---- head: concat [x | h] then 5 improved-GCN layers (dn2, fill=2) ----
  k_copycols<<<gcount((long)V.n * 3), 256, 0, stream>>>(V.x, 3, V.CAT, 208, 0, 3, V.n);
  k_copycols<<<gcount((long)V.n * 192), 256, 0, stream>>>(V.HOUT, 192, V.CAT, 208, 3, 192, V.n);
  gcn_layer(V, V.CAT, 208, 195, pc1, V.dn2, 2.0f, ACT_TANH, V.P1);   // [n,256]
  gcn_layer(V, V.P1, 256, 256, pc2, V.dn2, 2.0f, ACT_TANH, V.P2);    // [n,128]
  gcn_layer(V, V.P2, 128, 128, pc3, V.dn2, 2.0f, ACT_TANH, V.P1);    // [n,64]
  gcn_layer(V, V.P1, 64, 64, pc4, V.dn2, 2.0f, ACT_TANH, V.P2);      // [n,32]
  gcn_layer(V, V.P2, 32, 32, pco, V.dn2, 2.0f, ACT_NONE, V.P1);      // [n,3] (stride 16)
  k_strip<<<gcount((long)V.n * 3), 256, 0, stream>>>(V.P1, 16, (float*)d_out, V.n, 3);
}
